// DSDMSR_15075335209757
// MI455X (gfx1250) — matrix operand order repaired and checked by reading, not compiled, not run on hardware
//
#include <hip/hip_runtime.h>
#include <hip/hip_bf16.h>

typedef __attribute__((ext_vector_type(16))) _Float16 v16h;
typedef __attribute__((ext_vector_type(8)))  _Float16 v8h;
typedef __attribute__((ext_vector_type(8)))  float    v8f;

__device__ __forceinline__ v16h cat8(v8h x, v8h y) {
    return __builtin_shufflevector(x, y, 0,1,2,3,4,5,6,7,8,9,10,11,12,13,14,15);
}

__device__ __forceinline__ int clampi(int v, int lo, int hi) {
    return v < lo ? lo : (v > hi ? hi : v);
}

// half-pixel bilinear upsample sample: src is HxH, scale s, dst coord (y,x) in [0, s*H)
__device__ __forceinline__ float bil(const float* __restrict__ p, int H, int s, int y, int x) {
    float fy = (y + 0.5f) / (float)s - 0.5f;
    float fx = (x + 0.5f) / (float)s - 0.5f;
    int y0 = (int)floorf(fy), x0 = (int)floorf(fx);
    float wy = fy - (float)y0, wx = fx - (float)x0;
    int y1 = clampi(y0 + 1, 0, H - 1), x1 = clampi(x0 + 1, 0, H - 1);
    y0 = clampi(y0, 0, H - 1); x0 = clampi(x0, 0, H - 1);
    float a = p[y0 * H + x0], b = p[y0 * H + x1];
    float c = p[y1 * H + x0], d = p[y1 * H + x1];
    return a * (1.f - wy) * (1.f - wx) + b * (1.f - wy) * wx
         + c * wy * (1.f - wx) + d * wy * wx;
}

// LDS layout (dynamic):
//   s_in : float[33*48 ->1600] input tile, halo 8, zero apron (conv1 K padded to 16-tap rows)
//   s_w1 : f16[64*160]         conv1 weights [chan][k'], k'=dy*16+dx, zeros for dx>=9||dy>=9
//   s_w2 : f16[32*25*64]       conv2 weights [cout][tap][cin]
//   s_w3 : f16[25*32]          conv3 weights [tap][cin]
//   s_f1 : f16[24*24*64]       conv1 output (relu) [pix][chan]
//   s_f2 : f16[20*20*32]       conv2 output (relu) [pix][chan]
#define SIN_FLOATS 1600
#define SMEM_BYTES (SIN_FLOATS*4 + 2*(64*160 + 32*25*64 + 25*32 + 24*24*64 + 20*20*32))

__global__ void __launch_bounds__(256)
dsdmsr_unit(const float* __restrict__ in,
            const float* __restrict__ w1g, const float* __restrict__ b1g,
            const float* __restrict__ w2g, const float* __restrict__ b2g,
            const float* __restrict__ w3g, const float* __restrict__ b3g,
            int unit, int H, int W,
            float* __restrict__ outp, int outW, int oStride, int offY, int offX,
            int outBatchStride, int msf,
            const float* __restrict__ q2, const float* __restrict__ q4,
            const float* __restrict__ q8, const float* __restrict__ q16)
{
    extern __shared__ char smem[];
    float*    s_in = (float*)smem;                        // 1600 floats (33x48 used)
    _Float16* s_w1 = (_Float16*)(smem + SIN_FLOATS * 4);  // 10240
    _Float16* s_w2 = s_w1 + 64 * 160;                     // 51200
    _Float16* s_w3 = s_w2 + 32 * 25 * 64;                 // 800
    _Float16* s_f1 = s_w3 + 25 * 32;                      // 36864
    _Float16* s_f2 = s_f1 + 24 * 24 * 64;                 // 12800

    const int tid  = threadIdx.x;
    const int lane = tid & 31;
    const int wv   = tid >> 5;
    const int lhi  = lane >> 4;     // 0/1: lane half (K group select)
    const int lln  = lane & 15;
    const int b    = blockIdx.z;
    const int ty0  = blockIdx.y * 16;
    const int tx0  = blockIdx.x * 16;

    const float* inB = in ? (in + (size_t)b * H * W) : (const float*)0;

    // ---------------- Phase 0a: zero conv1 weight pad, coalesced w2/w3, input ---
    for (int i = tid; i < 64 * 160; i += 256) s_w1[i] = (_Float16)0.0f;
    // w2: read global fully coalesced in native (co,cin,tap) order, scatter into LDS
    for (int i = tid; i < 32 * 25 * 64; i += 256) {
        int co = i / 1600; int r = i - co * 1600; int cin = r / 25; int tap = r - cin * 25;
        s_w2[(co * 25 + tap) * 64 + cin] = (_Float16)w2g[unit * 51200 + i];
    }
    for (int i = tid; i < 800; i += 256) {
        int cin = i / 25, tap = i - cin * 25;
        s_w3[tap * 32 + cin] = (_Float16)w3g[unit * 800 + i];
    }
    for (int i = tid; i < 33 * 48; i += 256) {
        int ly = i / 48, lx = i % 48;
        int gy = ty0 - 8 + ly, gx = tx0 - 8 + lx;
        float v = 0.0f;
        if (gy >= 0 && gy < H && gx >= 0 && gx < W) {
            if (!msf) {
                v = inB[gy * W + gx];
            } else {
                v = bil(q2 + (size_t)b * 128 * 128, 128, 8, gy, gx)
                  + bil(q4 + (size_t)b * 256 * 256, 256, 4, gy, gx)
                  + bil(q8 + (size_t)b * 512 * 512, 512, 2, gy, gx)
                  + q16[(size_t)b * 1024 * 1024 + gy * 1024 + gx];
            }
        }
        s_in[i] = v;
    }
    __syncthreads();
    // ---------------- Phase 0b: coalesced w1 scatter over the zero fill ---------
    for (int i = tid; i < 64 * 81; i += 256) {
        int ch = i / 81; int kk = i - ch * 81; int dy = kk / 9; int dx = kk - dy * 9;
        s_w1[ch * 160 + dy * 16 + dx] = (_Float16)w1g[unit * 64 * 81 + i];
    }
    __syncthreads();

    // ---------------- Phase 1: conv1 9x9 1->64 + relu (WMMA, K padded to 160) ---
    // 36 pixel tiles (24*24/16); each wave owns a tile + all 4 channel tiles so the
    // im2col A fragment is built once and reused 4x. k' = dy*16+dx -> per unrolled
    // k-step dy/dx are constants: A gather = paired ds_load with const offsets.
    for (int pt = wv; pt < 36; pt += 8) {
        int m   = pt * 16 + lln;
        int j1y = m / 24, j1x = m % 24;
        const float*    abase = s_in + j1y * 48 + j1x + lhi * 48;
        const _Float16* bb0   = s_w1 + (0 * 16 + lln) * 160 + (lhi << 4);
        const _Float16* bb1   = s_w1 + (1 * 16 + lln) * 160 + (lhi << 4);
        const _Float16* bb2   = s_w1 + (2 * 16 + lln) * 160 + (lhi << 4);
        const _Float16* bb3   = s_w1 + (3 * 16 + lln) * 160 + (lhi << 4);
        v8f acc0 = {}, acc1 = {}, acc2 = {}, acc3 = {};
        #pragma unroll
        for (int kt = 0; kt < 5; ++kt) {
            const int kb  = kt * 32;
            const int dyb = kb >> 4;
            v16h a;
            #pragma unroll
            for (int h = 0; h < 16; ++h) {
                a[h] = (_Float16)abase[dyb * 48 + h];
            }
            const v8h* bp0 = (const v8h*)(bb0 + kb);
            const v8h* bp1 = (const v8h*)(bb1 + kb);
            const v8h* bp2 = (const v8h*)(bb2 + kb);
            const v8h* bp3 = (const v8h*)(bb3 + kb);
            acc0 = __builtin_amdgcn_wmma_f32_16x16x32_f16(false, a, false, cat8(bp0[0], bp0[1]), (short)0, acc0, false, false);
            acc1 = __builtin_amdgcn_wmma_f32_16x16x32_f16(false, a, false, cat8(bp1[0], bp1[1]), (short)0, acc1, false, false);
            acc2 = __builtin_amdgcn_wmma_f32_16x16x32_f16(false, a, false, cat8(bp2[0], bp2[1]), (short)0, acc2, false, false);
            acc3 = __builtin_amdgcn_wmma_f32_16x16x32_f16(false, a, false, cat8(bp3[0], bp3[1]), (short)0, acc3, false, false);
        }
        #pragma unroll
        for (int ct = 0; ct < 4; ++ct) {
            v8f acc = (ct == 0) ? acc0 : (ct == 1) ? acc1 : (ct == 2) ? acc2 : acc3;
            int ch = ct * 16 + lln;
            float bias = b1g[unit * 64 + ch];
            #pragma unroll
            for (int v = 0; v < 8; ++v) {
                int m2 = pt * 16 + (lhi << 3) + v;
                s_f1[m2 * 64 + ch] = (_Float16)fmaxf(acc[v] + bias, 0.0f);
            }
        }
    }
    __syncthreads();

    // ---------------- Phase 2: conv2 5x5 64->32 + relu (WMMA, K=1600) -----------
    // 25 pixel tiles (20*20/16) -> 13 tile-pairs; each wave owns 2 pixel tiles x
    // 2 channel tiles (4 accumulators): per k-step 8 ds_load_b128 feed 4 WMMAs
    // (2 loads/wmma), and weight fragments are shared across both pixel tiles.
    for (int p = wv; p < 13; p += 8) {
        int ptA = 2 * p;
        int ptB = (2 * p + 1 < 25) ? (2 * p + 1) : 24;   // clamp: dup work, no OOB
        int mA  = ptA * 16 + lln;
        int mB  = ptB * 16 + lln;
        int jAy = mA / 20, jAx = mA % 20;
        int jBy = mB / 20, jBx = mB % 20;
        const _Float16* abaseA = s_f1 + (jAy * 24 + jAx) * 64 + (lhi << 3);
        const _Float16* abaseB = s_f1 + (jBy * 24 + jBx) * 64 + (lhi << 3);
        const _Float16* bbas0  = s_w2 + (0 * 16 + lln) * 25 * 64 + (lhi << 3);
        const _Float16* bbas1  = s_w2 + (1 * 16 + lln) * 25 * 64 + (lhi << 3);
        v8f acc00 = {}, acc01 = {}, acc10 = {}, acc11 = {};
        #pragma unroll
        for (int kt = 0; kt < 50; ++kt) {
            const int tap = kt >> 1;
            const int c0  = (kt & 1) << 5;            // k = tap*64 + c (channel-minor)
            const int dy = tap / 5, dx = tap % 5;     // compile-time after unroll
            const int aoff = (dy * 24 + dx) * 64 + c0;
            const v8h* apA = (const v8h*)(abaseA + aoff);
            const v8h* apB = (const v8h*)(abaseB + aoff);
            v16h a0 = cat8(apA[0], apA[2]);           // +0 and +16 halves
            v16h a1 = cat8(apB[0], apB[2]);
            const v8h* bp0 = (const v8h*)(bbas0 + (tap * 64 + c0));
            const v8h* bp1 = (const v8h*)(bbas1 + (tap * 64 + c0));
            v16h b0 = cat8(bp0[0], bp0[2]);
            v16h b1 = cat8(bp1[0], bp1[2]);
            acc00 = __builtin_amdgcn_wmma_f32_16x16x32_f16(false, a0, false, b0, (short)0, acc00, false, false);
            acc01 = __builtin_amdgcn_wmma_f32_16x16x32_f16(false, a0, false, b1, (short)0, acc01, false, false);
            acc10 = __builtin_amdgcn_wmma_f32_16x16x32_f16(false, a1, false, b0, (short)0, acc10, false, false);
            acc11 = __builtin_amdgcn_wmma_f32_16x16x32_f16(false, a1, false, b1, (short)0, acc11, false, false);
        }
        #pragma unroll
        for (int q = 0; q < 4; ++q) {
            v8f acc = (q == 0) ? acc00 : (q == 1) ? acc01 : (q == 2) ? acc10 : acc11;
            int pt  = (q < 2) ? ptA : ptB;
            int co  = ((q & 1) ? 16 : 0) + lln;
            float bias = b2g[unit * 32 + co];
            #pragma unroll
            for (int v = 0; v < 8; ++v) {
                int m2 = pt * 16 + (lhi << 3) + v;
                s_f2[m2 * 32 + co] = (_Float16)fmaxf(acc[v] + bias, 0.0f);
            }
        }
    }
    __syncthreads();

    // ---------------- Phase 3: conv3 5x5 32->1 + interleaved store --------------
    {
        int oy = tid >> 4, ox = tid & 15;
        float sum = b3g[unit];
        for (int dy = 0; dy < 5; ++dy) {
            for (int dx = 0; dx < 5; ++dx) {
                const _Float16* f = s_f2 + ((oy + dy) * 20 + (ox + dx)) * 32;
                const _Float16* w = s_w3 + (dy * 5 + dx) * 32;
                #pragma unroll
                for (int c = 0; c < 32; ++c)
                    sum += (float)f[c] * (float)w[c];
            }
        }
        int gy = ty0 + oy, gx = tx0 + ox;
        size_t oidx = (size_t)b * outBatchStride
                    + (size_t)(gy * oStride + offY) * outW + (size_t)(gx * oStride + offX);
        outp[oidx] = sum;
    }
}

extern "C" void kernel_launch(void* const* d_in, const int* in_sizes, int n_in,
                              void* d_out, int out_size, void* d_ws, size_t ws_size,
                              hipStream_t stream) {
    (void)in_sizes; (void)n_in; (void)out_size; (void)d_ws; (void)ws_size;

    const float* image = (const float*)d_in[0];
    const float* w1 = (const float*)d_in[1];
    const float* b1 = (const float*)d_in[2];
    const float* w2 = (const float*)d_in[3];
    const float* b2 = (const float*)d_in[4];
    const float* w3 = (const float*)d_in[5];
    const float* b3 = (const float*)d_in[6];

    float* out  = (float*)d_out;
    float* o2   = out;                       // (2,1,128,128)
    float* o4   = o2  + 2 * 128 * 128;       // (2,1,256,256)
    float* o8   = o4  + 2 * 256 * 256;       // (2,1,512,512)
    float* o16  = o8  + 2 * 512 * 512;       // (2,1,1024,1024)
    float* omsf = o16 + 2 * 1024 * 1024;     // (2,1,1024,1024)

    const size_t shmem = (size_t)SMEM_BYTES;
    (void)hipFuncSetAttribute((const void*)dsdmsr_unit,
                              hipFuncAttributeMaxDynamicSharedMemorySize, (int)shmem);

    auto launch_stage = [&](const float* in, int H, int ubase, float* o) {
        dim3 grid(H / 16, H / 16, 2);
        for (int j = 0; j < 4; ++j) {
            dsdmsr_unit<<<grid, 256, shmem, stream>>>(
                in, w1, b1, w2, b2, w3, b3, ubase + j, H, H,
                o, 2 * H, /*oStride=*/2, /*offY=*/(j >> 1), /*offX=*/(j & 1),
                /*outBatchStride=*/4 * H * H, /*msf=*/0,
                nullptr, nullptr, nullptr, nullptr);
        }
    };

    launch_stage(image, 64,  0,  o2);
    launch_stage(o2,    128, 4,  o4);
    launch_stage(o4,    256, 8,  o8);
    launch_stage(o8,    512, 12, o16);

    // MSF head: input = upscale(o2,8)+upscale(o4,4)+upscale(o8,2)+o16, built on the fly
    dsdmsr_unit<<<dim3(64, 64, 2), 256, shmem, stream>>>(
        nullptr, w1, b1, w2, b2, w3, b3, 16, 1024, 1024,
        omsf, 1024, /*oStride=*/1, 0, 0,
        /*outBatchStride=*/1024 * 1024, /*msf=*/1,
        o2, o4, o8, o16);
}
